// QRNN_22771916603950
// MI455X (gfx1250) — matrix operand order repaired and checked by reading, not compiled, not run on hardware
//
#include <hip/hip_runtime.h>
#include <hip/hip_bf16.h>
#include <math.h>

// ---------------------------------------------------------------------------
// QRNN: wh_out = x @ Wh + b (big bf16 WMMA GEMM), then 512 sequential steps
// h_t = tanh(wh_out_t + h_{t-1} @ Uh) in a persistent kernel with a grid
// barrier. v2: Uh^T block slice staged ONCE in LDS (132KB dynamic, padded to
// kill bank conflicts) and the h_prev A-fragment loads software-pipelined.
// ---------------------------------------------------------------------------

typedef __bf16 bf16_t;
typedef __attribute__((ext_vector_type(16))) __bf16 v16bf;
typedef __attribute__((ext_vector_type(8)))  float  v8f;

#define TSTEPS 512
#define BATCH  32
#define DIM    1024
#define QDIM   256

// workspace byte offsets
#define WS_WHT   ((size_t)0)                       // D*D bf16  = 2 MB
#define WS_UHT   ((size_t)(2u*1024u*1024u))        // D*D bf16  = 2 MB
#define WS_WHOUT ((size_t)(4u*1024u*1024u))        // T*B*D f32 = 64 MB
#define WS_HBUF0 ((size_t)(68u*1024u*1024u))       // B*D bf16  = 64 KB
#define WS_HBUF1 (WS_HBUF0 + (size_t)(64u*1024u))  // B*D bf16  = 64 KB
#define WS_CTR   (WS_HBUF0 + (size_t)(128u*1024u)) // grid-barrier counter

// ---------------- WMMA fragment helpers (wave32, 16x16x32 bf16) ------------

__device__ __forceinline__ int a_kpair(int p, int g) {
    // A 16x32: VGPR v holds K={2v,2v+1}(+16 for lanes 16-31)
    return (g << 4) + (p << 1);
}

__device__ __forceinline__ v16bf load_a_from_f32(const float* __restrict__ base,
                                                 int ld, int lane) {
    const int g = lane >> 4, m = lane & 15;
    const float* row = base + (size_t)m * ld;
    v16bf a;
#pragma unroll
    for (int p = 0; p < 8; ++p) {
        const float2 v = *(const float2*)(row + a_kpair(p, g));
        a[2*p]   = (bf16_t)v.x;
        a[2*p+1] = (bf16_t)v.y;
    }
    return a;
}

__device__ __forceinline__ v16bf load_a_from_bf16(const bf16_t* __restrict__ base,
                                                  int ld, int lane) {
    const int g = lane >> 4, m = lane & 15;
    const bf16_t* row = base + (size_t)m * ld;
    v16bf a;
#pragma unroll
    for (int p = 0; p < 8; ++p) {
        union { unsigned u; bf16_t h[2]; } w;
        w.u = *(const unsigned*)(row + a_kpair(p, g));
        a[2*p]   = w.h[0];
        a[2*p+1] = w.h[1];
    }
    return a;
}

struct B32Bytes { uint4 lo, hi; };

// B 32x16 from bf16 source stored transposed [n][k]: lane's 16 K-values
// (K = e + 16*g) are contiguous -> two 16B loads.
__device__ __forceinline__ v16bf load_b_bf16(const bf16_t* __restrict__ base,
                                             int ld, int lane) {
    const int g = lane >> 4, n = lane & 15;
    const bf16_t* p = base + (size_t)n * ld + (g << 4);
    B32Bytes raw;
    raw.lo = *(const uint4*)(p);
    raw.hi = *(const uint4*)(p + 8);
    return __builtin_bit_cast(v16bf, raw);
}

// Same fragment shape, but from an LDS-resident padded slice.
__device__ __forceinline__ v16bf load_b_from_lds(const bf16_t* lds, int idx) {
    B32Bytes raw;
    raw.lo = *(const uint4*)(lds + idx);
    raw.hi = *(const uint4*)(lds + idx + 8);
    return __builtin_bit_cast(v16bf, raw);
}

__device__ __forceinline__ v8f wmma_bf16(v16bf a, v16bf b, v8f c) {
    return __builtin_amdgcn_wmma_f32_16x16x32_bf16(
        /*neg_a=*/false, a, /*neg_b=*/false, b,
        /*c_mod=*/(short)0, c, /*reuse_a=*/false, /*reuse_b=*/false);
}

// ---------------- Kernel 1: build WhT / UhT (bf16, transposed) -------------
// Wh[d,e] = sign[d/Q][e/Q] * comp[d/Q][e/Q][d%Q, e%Q]; we store WhT[e*D+d].

__global__ __launch_bounds__(256) void build_qmat_kernel(
    const float* __restrict__ wr, const float* __restrict__ wi,
    const float* __restrict__ wj, const float* __restrict__ wk,
    const float* __restrict__ ur, const float* __restrict__ ui,
    const float* __restrict__ uj, const float* __restrict__ uk,
    bf16_t* __restrict__ whT, bf16_t* __restrict__ uhT) {
    unsigned idx = blockIdx.x * 256u + threadIdx.x;          // 0 .. 2*D*D-1
    const unsigned DD = (unsigned)DIM * (unsigned)DIM;
    const bool isU = idx >= DD;
    const unsigned li = isU ? (idx - DD) : idx;
    const int e = (int)(li / DIM), d = (int)(li % DIM);
    const int dr = d >> 8, ec = e >> 8;                      // quaternion block
    const int   comp[4][4] = {{0,1,2,3},{1,0,3,2},{2,3,0,1},{3,2,1,0}};
    const float sgn [4][4] = {{ 1, 1, 1, 1},{-1, 1, 1,-1},
                              {-1,-1, 1, 1},{-1, 1,-1, 1}};
    const float* srcs[8] = {wr, wi, wj, wk, ur, ui, uj, uk};
    const float* s = srcs[comp[dr][ec] + (isU ? 4 : 0)];
    const float v = sgn[dr][ec] * s[(size_t)(d & 255) * QDIM + (e & 255)];
    (isU ? uhT : whT)[li] = (bf16_t)v;
}

// ---------------- Kernel 2: init h0 = 0 and barrier counter = 0 ------------

__global__ __launch_bounds__(256) void init_state_kernel(unsigned* __restrict__ h0,
                                                         unsigned* __restrict__ ctr,
                                                         int n) {
    const int i = blockIdx.x * 256 + threadIdx.x;
    if (i < n) h0[i] = 0u;
    if (i == 0) *ctr = 0u;
}

// ---------------- Kernel 3: wh_out = x @ Wh + b (bf16 WMMA) ----------------
// grid = (M/16, 4), block = 128 (4 waves). Each wave: 16 rows x 64 cols.
// Throughput-bound phase: occupancy hides L2 latency, no LDS needed.

__global__ __launch_bounds__(128) void gemm_wh_kernel(
    const float* __restrict__ x, const bf16_t* __restrict__ whT,
    const float* __restrict__ bias, float* __restrict__ whout) {
    const int wave = threadIdx.x >> 5, lane = threadIdx.x & 31;
    const int m0 = blockIdx.x * 16;
    const int n0 = blockIdx.y * 256 + wave * 64;

    v8f acc[4] = {v8f{}, v8f{}, v8f{}, v8f{}};
    for (int k0 = 0; k0 < DIM; k0 += 32) {
        const v16bf a = load_a_from_f32(x + (size_t)m0 * DIM + k0, DIM, lane);
#pragma unroll
        for (int s = 0; s < 4; ++s) {
            const v16bf b =
                load_b_bf16(whT + (size_t)(n0 + 16 * s) * DIM + k0, DIM, lane);
            acc[s] = wmma_bf16(a, b, acc[s]);
        }
    }
    const int g = lane >> 4, n = lane & 15;
#pragma unroll
    for (int s = 0; s < 4; ++s) {
        const int col = n0 + 16 * s + n;
        const float bv = bias[col];
#pragma unroll
        for (int v = 0; v < 8; ++v) {
            const int row = m0 + v + 8 * g;
            whout[(size_t)row * DIM + col] = acc[s][v] + bv;
        }
    }
}

// ---------------- Kernel 4: persistent recurrence --------------------------
// 16 blocks x 128 thr = 64 waves. Block b owns cols [64b, 64b+64); waves 0/1
// handle local cols 0..31 for batch rows 0..15 / 16..31, waves 2/3 local
// cols 32..63. Uh^T slice (64 x DIM bf16) staged once in padded LDS; the
// per-step h_prev A-fragments are prefetched one K-chunk ahead.
// Grid barrier = monotonic atomic counter (reset each launch).

#define RECUR_BLOCKS 16
#define BSTRIDE (DIM + 8)   // +16B pad: lane stride 2064B == 16 mod 256 ->
                            // the 16 B-fragment lanes cover all 64 LDS banks
#define RECUR_LDS_BYTES ((size_t)(64 * BSTRIDE * sizeof(bf16_t)))  // 132,096 B

__global__ __launch_bounds__(128) void qrnn_recur_kernel(
    const float* __restrict__ whout, const bf16_t* __restrict__ uhT,
    float* __restrict__ hout, bf16_t* __restrict__ hb0,
    bf16_t* __restrict__ hb1, unsigned* __restrict__ ctr) {
    extern __shared__ bf16_t lds_b[];   // 64 rows x BSTRIDE bf16

    const int wave = threadIdx.x >> 5, lane = threadIdx.x & 31;
    const int m0 = (wave & 1) * 16;             // batch-row tile
    const int nloc = (wave >> 1) * 32;          // local col tile (0 or 32)
    const int colbase = blockIdx.x * 64;
    const int g = lane >> 4, n = lane & 15;

    // --- one-time stage: Uh^T[colbase .. colbase+63][*] -> LDS (coalesced) --
    for (int c = threadIdx.x; c < 64 * (DIM / 8); c += 128) {
        const int r = c >> 7;              // row 0..63   (DIM/8 == 128)
        const int k = (c & 127) << 3;      // element 0..1016 step 8
        *(uint4*)&lds_b[r * BSTRIDE + k] =
            *(const uint4*)&uhT[(size_t)(colbase + r) * DIM + k];
    }
    __syncthreads();

    const int brow0 = (nloc + n) * BSTRIDE + (g << 4);
    const int brow1 = (nloc + 16 + n) * BSTRIDE + (g << 4);

    for (int t = 0; t < TSTEPS; ++t) {
        const bf16_t* __restrict__ hprev = (t & 1) ? hb1 : hb0;
        bf16_t* __restrict__ hcur        = (t & 1) ? hb0 : hb1;
        const bf16_t* __restrict__ arow  = hprev + (size_t)m0 * DIM;

        v8f acc0 = v8f{}, acc1 = v8f{};
        v16bf a = load_a_from_bf16(arow, DIM, lane);          // K-chunk 0
        for (int k0 = 0; k0 < DIM; k0 += 32) {
            v16bf a_next = a;
            if (k0 + 32 < DIM)                                // prefetch next
                a_next = load_a_from_bf16(arow + k0 + 32, DIM, lane);
            const v16bf b0 = load_b_from_lds(lds_b, brow0 + k0);
            const v16bf b1 = load_b_from_lds(lds_b, brow1 + k0);
            acc0 = wmma_bf16(a, b0, acc0);
            acc1 = wmma_bf16(a, b1, acc1);
            a = a_next;
        }

        const float* __restrict__ wo = whout + (size_t)t * BATCH * DIM;
        float* __restrict__ ho       = hout  + (size_t)t * BATCH * DIM;
        const int c0 = colbase + nloc + n;
        const int c1 = c0 + 16;
#pragma unroll
        for (int v = 0; v < 8; ++v) {
            const int row = m0 + v + 8 * g;
            const float h0 = tanhf(acc0[v] + wo[(size_t)row * DIM + c0]);
            const float h1 = tanhf(acc1[v] + wo[(size_t)row * DIM + c1]);
            ho[(size_t)row * DIM + c0] = h0;
            ho[(size_t)row * DIM + c1] = h1;
            hcur[(size_t)row * DIM + c0] = (bf16_t)h0;
            hcur[(size_t)row * DIM + c1] = (bf16_t)h1;
        }

        if (t != TSTEPS - 1) {
            __threadfence();          // release this block's h_t writes
            __syncthreads();
            if (threadIdx.x == 0) {
                __hip_atomic_fetch_add(ctr, 1u, __ATOMIC_RELEASE,
                                       __HIP_MEMORY_SCOPE_AGENT);
                const unsigned target = (unsigned)RECUR_BLOCKS * (unsigned)(t + 1);
                while (__hip_atomic_load(ctr, __ATOMIC_ACQUIRE,
                                         __HIP_MEMORY_SCOPE_AGENT) < target) {
                    __builtin_amdgcn_s_sleep(1);
                }
            }
            __syncthreads();
            __threadfence();          // acquire other blocks' h_t writes
        }
    }
}

// ---------------------------------------------------------------------------

extern "C" void kernel_launch(void* const* d_in, const int* in_sizes, int n_in,
                              void* d_out, int out_size, void* d_ws, size_t ws_size,
                              hipStream_t stream) {
    (void)in_sizes; (void)n_in; (void)out_size; (void)ws_size;

    const float* x    = (const float*)d_in[0];
    const float* wh_r = (const float*)d_in[1];
    const float* wh_i = (const float*)d_in[2];
    const float* wh_j = (const float*)d_in[3];
    const float* wh_k = (const float*)d_in[4];
    const float* uh_r = (const float*)d_in[5];
    const float* uh_i = (const float*)d_in[6];
    const float* uh_j = (const float*)d_in[7];
    const float* uh_k = (const float*)d_in[8];
    const float* wh_b = (const float*)d_in[9];
    float* out = (float*)d_out;

    char* ws = (char*)d_ws;
    bf16_t*   whT   = (bf16_t*)(ws + WS_WHT);
    bf16_t*   uhT   = (bf16_t*)(ws + WS_UHT);
    float*    whout = (float*)(ws + WS_WHOUT);
    bf16_t*   hb0   = (bf16_t*)(ws + WS_HBUF0);
    bf16_t*   hb1   = (bf16_t*)(ws + WS_HBUF1);
    unsigned* ctr   = (unsigned*)(ws + WS_CTR);

    // 1) build quaternion matrices (bf16, transposed for B-fragment loads)
    {
        const int total = 2 * DIM * DIM;
        build_qmat_kernel<<<(total + 255) / 256, 256, 0, stream>>>(
            wh_r, wh_i, wh_j, wh_k, uh_r, uh_i, uh_j, uh_k, whT, uhT);
    }
    // 2) zero h0 and the grid-barrier counter (deterministic per launch)
    {
        const int n = BATCH * DIM / 2;  // bf16 pairs as u32
        init_state_kernel<<<(n + 256) / 256, 256, 0, stream>>>((unsigned*)hb0, ctr, n);
    }
    // 3) wh_out = x @ Wh + b   (16384 x 1024 x 1024, bf16 WMMA)
    {
        dim3 grid(TSTEPS * BATCH / 16, 4);
        gemm_wh_kernel<<<grid, 128, 0, stream>>>(x, whT, wh_b, whout);
    }
    // 4) persistent recurrence over 512 steps (132KB dynamic LDS per block)
    qrnn_recur_kernel<<<RECUR_BLOCKS, 128, RECUR_LDS_BYTES, stream>>>(
        whout, uhT, out, hb0, hb1, ctr);
}
